// ContextMemory_33904471835543
// MI455X (gfx1250) — matrix operand order repaired and checked by reading, not compiled, not run on hardware
//
#include <hip/hip_runtime.h>
#include <hip/hip_bf16.h>

// ---------------------------------------------------------------------------
// ContextMemory for MI455X (gfx1250, wave32, WMMA + TDM).
// Heavy GEMMs: v_wmma_f32_16x16x32_bf16, fp32 accumulation, BK=64 stages.
// Tile staging of bf16 operands uses the Tensor Data Mover
// (tensor_load_to_lds + s_wait_tensorcnt).  Second MHA collapses
// analytically (softmax over 1 key == 1.0).
// ---------------------------------------------------------------------------

typedef __attribute__((ext_vector_type(16))) __bf16 bf16x16;
typedef __attribute__((ext_vector_type(8)))  __bf16 bf16x8;
typedef __attribute__((ext_vector_type(4)))  __bf16 bf16x4;
typedef __attribute__((ext_vector_type(8)))  float  f32x8;
typedef __attribute__((ext_vector_type(4)))  unsigned u32x4;
typedef __attribute__((ext_vector_type(8)))  int      i32x8;
typedef __attribute__((ext_vector_type(4)))  int      i32x4;

#define Bsz   8
#define Ssz   2048
#define Msz   512
#define Esz   1024
#define Hsz   16
#define Dsz   64
#define LNEPS 1e-5f

// ---- WMMA helpers ---------------------------------------------------------

__device__ __attribute__((always_inline)) inline
f32x8 wmma_bf16(bf16x16 a, bf16x16 b, f32x8 c) {
  return __builtin_amdgcn_wmma_f32_16x16x32_bf16(
      /*neg_a=*/false, a, /*neg_b=*/false, b,
      /*c_mod=*/(short)0, c, /*reuse_a=*/false, /*reuse_b=*/false);
}

// A fragment (16x32, MxK).  Lane: m = lane&15, h = lane>>4.
// Per ISA: lane holds K = {h*8 .. h*8+7} and {16+h*8 .. 16+h*8+7}.
__device__ __attribute__((always_inline)) inline
bf16x16 load_frag_a(const __bf16* base, int ld) {
  const int lane = threadIdx.x & 31;
  const int m = lane & 15, h = lane >> 4;
  const __bf16* p = base + m * ld + h * 8;
  bf16x8 lo = *(const bf16x8*)(p);
  bf16x8 hi = *(const bf16x8*)(p + 16);
  bf16x16 f;
#pragma unroll
  for (int i = 0; i < 8; ++i) { f[i] = lo[i]; f[8 + i] = hi[i]; }
  return f;
}

// B fragment (32x16, KxN) from row-major (N x K) storage (i.e. B^T rows).
// Same K order as A: lane holds K = {h*8 .. h*8+7} and {16+h*8 .. 16+h*8+7}.
__device__ __attribute__((always_inline)) inline
bf16x16 load_frag_b(const __bf16* base, int ld) {
  const int lane = threadIdx.x & 31;
  const int n = lane & 15, h = lane >> 4;
  const __bf16* p = base + n * ld + h * 8;
  bf16x8 lo = *(const bf16x8*)(p);
  bf16x8 hi = *(const bf16x8*)(p + 16);
  bf16x16 f;
#pragma unroll
  for (int i = 0; i < 8; ++i) { f[i] = lo[i]; f[8 + i] = hi[i]; }
  return f;
}

// ---- Tensor Data Mover: 2-D bf16 tile  global -> LDS ----------------------
// D# built per CDNA5 ISA ch.8: group0 {count, lds_addr, global_addr, type=2},
// group1 {data_size=2B, tensor_dim0/1, tile_dim0/1, dim0 stride}.
// Must be issued from a single wave per workgroup (EXEC is ignored);
// completion via s_wait_tensorcnt in the issuing wave, then s_barrier.

__device__ __attribute__((always_inline)) inline
unsigned lds_offset(const void* p) {            // LDS generic addr: low 32b = offset
  return (unsigned)(unsigned long long)p;
}

__device__ __attribute__((always_inline)) inline
void tdm_load_2d_bf16(void* lds_dst, const __bf16* gsrc,
                      unsigned rows, unsigned row_elems, unsigned stride_elems) {
  unsigned long long ga = (unsigned long long)gsrc;
  u32x4 g0;
  g0[0] = 1u;                                            // count=1, user D#
  g0[1] = lds_offset(lds_dst);                           // lds_addr (bytes)
  g0[2] = (unsigned)(ga & 0xffffffffu);                  // global_addr[31:0]
  g0[3] = (unsigned)((ga >> 32) & 0x01ffffffu) | (2u << 30);  // [56:32] | type=2
  i32x8 g1;
  g1[0] = (int)(1u << 16);                               // data_size=1 (2 bytes)
  g1[1] = (int)((row_elems & 0xffffu) << 16);            // tensor_dim0[15:0]
  g1[2] = (int)(((row_elems >> 16) & 0xffffu) |          // tensor_dim0[31:16]
                ((rows & 0xffffu) << 16));               // tensor_dim1[15:0]
  g1[3] = (int)(((rows >> 16) & 0xffffu) |               // tensor_dim1[31:16]
                ((row_elems & 0xffffu) << 16));          // tile_dim0
  g1[4] = (int)(rows & 0xffffu);                         // tile_dim1 (tile_dim2=0)
  g1[5] = (int)stride_elems;                             // tensor_dim0_stride[31:0]
  g1[6] = 0;
  g1[7] = 0;
  i32x4 z4 = {};
#if defined(__clang_major__) && __clang_major__ >= 23
  i32x8 z8 = {};
  __builtin_amdgcn_tensor_load_to_lds(g0, g1, z4, z4, z8, 0);
#else
  __builtin_amdgcn_tensor_load_to_lds(g0, g1, z4, z4, 0);
#endif
}

__device__ __attribute__((always_inline)) inline
void tdm_wait_all() { __builtin_amdgcn_s_wait_tensorcnt(0); }

// ---- small kernels --------------------------------------------------------

__global__ void k_convert(const float* __restrict__ src, __bf16* __restrict__ dst,
                          int cols, int src_ld, int total) {
  int idx = blockIdx.x * blockDim.x + threadIdx.x;
  if (idx >= total) return;
  int r = idx / cols, c = idx - r * cols;
  dst[idx] = (__bf16)src[(size_t)r * src_ld + c];
}

__global__ void k_wq(const float* __restrict__ q, float* __restrict__ wq) {
  int idx = blockIdx.x * blockDim.x + threadIdx.x;   // B*E threads
  int b = idx >> 10, e = idx & 1023;
  const float* p = q + (size_t)b * Ssz * Esz + e;
  float s = 0.f;
  for (int i = 0; i < Ssz; ++i) s += p[(size_t)i * Esz];
  wq[idx] = s * (1.f / (float)Ssz);
}

__global__ void k_gemv(const float* __restrict__ x, const float* __restrict__ W,
                       int ldw, const float* __restrict__ bias,
                       float* __restrict__ y, int NB, int N, int K) {
  int idx = blockIdx.x * blockDim.x + threadIdx.x;
  if (idx >= NB * N) return;
  int b = idx / N, n = idx - b * N;
  const float* xr = x + (size_t)b * K;
  const float* wr = W + (size_t)n * ldw;
  float s = bias ? bias[n] : 0.f;
  for (int k = 0; k < K; ++k) s += xr[k] * wr[k];
  y[idx] = s;
}

// ---- WMMA GEMM:  C[M x N] = (A[M x K] @ W[N x K]^T + bias) * scale --------
// Block tile 128x64, 8 waves of 32x32, BK=64 (two WMMA k-steps per stage).
// W tile staged by the TDM; A tile staged manually (fp32->bf16 convert path).

template <typename AT, bool OUT_BF16>
__global__ __launch_bounds__(256)
void k_gemm(const AT* __restrict__ A, int lda,
            const __bf16* __restrict__ W, int ldw,
            const float* __restrict__ bias,
            void* __restrict__ Cout, int ldc,
            int K, float scale) {
  __shared__ __align__(16) __bf16 As[128 * 64];
  __shared__ __align__(16) __bf16 Ws[64 * 64];

  const int tid  = threadIdx.x;
  const int r0   = blockIdx.y * 128;
  const int n0   = blockIdx.x * 64;
  const int wave = tid >> 5;
  const int wr   = wave >> 1;       // 0..3 -> 32-row sub-tile
  const int wc   = wave & 1;        // 0..1 -> 32-col sub-tile

  f32x8 acc[2][2] = {};

  for (int k0 = 0; k0 < K; k0 += 64) {
    __syncthreads();                 // previous stage fully consumed
    // ---- W tile (64 x 64) via Tensor Data Mover (wave 0 only) ----
    if (tid < 32)
      tdm_load_2d_bf16(Ws, &W[(size_t)n0 * ldw + k0], 64, 64, (unsigned)ldw);
    // ---- stage A tile (128 x 64) into LDS as bf16 ----
    if constexpr (sizeof(AT) == 4) {           // fp32 source: convert on stage
#pragma unroll
      for (int i = 0; i < 8; ++i) {
        int q = i * 256 + tid;                 // 2048 float4 chunks
        int row = q >> 4, cc = q & 15;
        float4 v = *(const float4*)&A[(size_t)(r0 + row) * lda + k0 + cc * 4];
        bf16x4 t; t[0] = (__bf16)v.x; t[1] = (__bf16)v.y;
        t[2] = (__bf16)v.z; t[3] = (__bf16)v.w;
        *(bf16x4*)&As[row * 64 + cc * 4] = t;
      }
    } else {                                   // bf16 source: straight copy
#pragma unroll
      for (int i = 0; i < 4; ++i) {
        int q = i * 256 + tid;                 // 1024 16B chunks
        int row = q >> 3, cc = q & 7;
        *(bf16x8*)&As[row * 64 + cc * 8] =
            *(const bf16x8*)&A[(size_t)(r0 + row) * lda + k0 + cc * 8];
      }
    }
    if (k0 + 64 < K)
      __builtin_prefetch(&A[(size_t)(r0 + (tid & 127)) * lda + k0 + 64], 0, 1);
    if (tid < 32) tdm_wait_all();    // TDM done before the barrier releases
    __syncthreads();

#pragma unroll
    for (int kk = 0; kk < 2; ++kk) {
      const int ko = kk * 32;
      bf16x16 a0 = load_frag_a(&As[(wr * 32 + 0)  * 64] + ko, 64);
      bf16x16 a1 = load_frag_a(&As[(wr * 32 + 16) * 64] + ko, 64);
      bf16x16 b0 = load_frag_b(&Ws[(wc * 32 + 0)  * 64] + ko, 64);
      bf16x16 b1 = load_frag_b(&Ws[(wc * 32 + 16) * 64] + ko, 64);
      acc[0][0] = wmma_bf16(a0, b0, acc[0][0]);
      acc[0][1] = wmma_bf16(a0, b1, acc[0][1]);
      acc[1][0] = wmma_bf16(a1, b0, acc[1][0]);
      acc[1][1] = wmma_bf16(a1, b1, acc[1][1]);
    }
  }

  // ---- epilogue: C layout lane n = lane&15, row = 8*(lane>>4) + vgpr ----
  const int lane = tid & 31, nl = lane & 15, hf = lane >> 4;
#pragma unroll
  for (int mi = 0; mi < 2; ++mi) {
#pragma unroll
    for (int ni = 0; ni < 2; ++ni) {
      int colg = n0 + wc * 32 + ni * 16 + nl;
      float bv = bias ? bias[colg] : 0.f;
#pragma unroll
      for (int r = 0; r < 8; ++r) {
        int rowg = r0 + wr * 32 + mi * 16 + hf * 8 + r;
        float v = (acc[mi][ni][r] + bv) * scale;
        if constexpr (OUT_BF16)
          ((__bf16*)Cout)[(size_t)rowg * ldc + colg] = (__bf16)v;
        else
          ((float*)Cout)[(size_t)rowg * ldc + colg] = v;
      }
    }
  }
}

// ---- flash attention over the 512-entry memory ----------------------------
// grid = (S/64, H, B), block = 128 (4 waves). Each wave owns 16 query rows.
// Qp is pre-scaled by 1/sqrt(D).  Q and K tiles staged by TDM.

__global__ __launch_bounds__(128)
void k_attn(const __bf16* __restrict__ Qp, const __bf16* __restrict__ Kb,
            const __bf16* __restrict__ Vb, __bf16* __restrict__ ctx) {
  __shared__ __align__(16) __bf16 Qs[64 * 64];
  __shared__ __align__(16) __bf16 Ks[64 * 64];
  __shared__ __align__(16) __bf16 Vt[64 * 64];   // V transposed: [d][key]
  __shared__ __align__(16) __bf16 Ps[64 * 64];   // softmax probs (bf16)
  __shared__ __align__(16) float  Ss[64 * 64];   // raw scores (fp32)
  __shared__ float m_s[64], l_s[64], a_s[64];

  const int tid  = threadIdx.x;
  const int qt   = blockIdx.x, h = blockIdx.y, b = blockIdx.z;
  const int hc   = h * Dsz;
  const size_t qrow0 = (size_t)b * Ssz + qt * 64;
  const int wave = tid >> 5, lane = tid & 31, nl = lane & 15, hf = lane >> 4;

  // Q tile (64 x 64) via TDM, overlapped with stats init
  if (tid < 32)
    tdm_load_2d_bf16(Qs, &Qp[qrow0 * Esz + hc], 64, 64, Esz);
  if (tid < 64) { m_s[tid] = -1e30f; l_s[tid] = 0.f; }

  f32x8 oc[4] = {};                        // 16 rows x 64 D accumulator

  for (int kb = 0; kb < Msz / 64; ++kb) {
    __syncthreads();
    // stage K block via TDM; V block transposed manually (TDM can't transpose)
    if (tid < 32)
      tdm_load_2d_bf16(Ks, &Kb[(size_t)(kb * 64) * Esz + hc], 64, 64, Esz);
#pragma unroll
    for (int i = 0; i < 4; ++i) {
      int q = i * 128 + tid;
      int row = q >> 3, cc = q & 7;
      bf16x8 v = *(const bf16x8*)&Vb[(size_t)(kb * 64 + row) * Esz + hc + cc * 8];
#pragma unroll
      for (int j = 0; j < 8; ++j) Vt[(cc * 8 + j) * 64 + row] = v[j];
    }
    if (tid < 32) tdm_wait_all();          // Q (first iter) + K in LDS
    __syncthreads();

    // scores: S = Q_tile(16x64) @ K_blk^T  -> 4 16x16 tiles, K-dim 64
    const __bf16* qbase = &Qs[(wave * 16) * 64];
#pragma unroll
    for (int nt = 0; nt < 4; ++nt) {
      f32x8 s = {};
#pragma unroll
      for (int kk = 0; kk < 2; ++kk) {
        bf16x16 af = load_frag_a(qbase + kk * 32, 64);
        bf16x16 bf = load_frag_b(&Ks[(nt * 16) * 64] + kk * 32, 64);
        s = wmma_bf16(af, bf, s);
      }
#pragma unroll
      for (int r = 0; r < 8; ++r)
        Ss[(wave * 16 + hf * 8 + r) * 64 + nt * 16 + nl] = s[r];
    }
    __syncthreads();

    // online softmax: 2 lanes per row, each scans 32 keys
    {
      const int rl = lane & 15, seg = lane >> 4;
      const int grow = wave * 16 + rl;
      float lm = -1e30f;
#pragma unroll
      for (int j = 0; j < 32; ++j)
        lm = fmaxf(lm, Ss[grow * 64 + seg * 32 + j]);
      lm = fmaxf(lm, __shfl_xor(lm, 16, 32));
      float mold = m_s[grow];
      float mnew = fmaxf(mold, lm);
      float alpha = __expf(mold - mnew);
      float ls = 0.f;
#pragma unroll
      for (int j = 0; j < 32; ++j) {
        float p = __expf(Ss[grow * 64 + seg * 32 + j] - mnew);
        Ps[grow * 64 + seg * 32 + j] = (__bf16)p;
        ls += p;
      }
      ls += __shfl_xor(ls, 16, 32);
      if (seg == 0) {
        m_s[grow] = mnew;
        l_s[grow] = l_s[grow] * alpha + ls;
        a_s[grow] = alpha;
      }
    }
    __syncthreads();

    // rescale running output by alpha, then O += P @ V
    float ar[8];
#pragma unroll
    for (int r = 0; r < 8; ++r) ar[r] = a_s[wave * 16 + hf * 8 + r];
#pragma unroll
    for (int ct = 0; ct < 4; ++ct)
#pragma unroll
      for (int r = 0; r < 8; ++r) oc[ct][r] *= ar[r];

    const __bf16* pbase = &Ps[(wave * 16) * 64];
#pragma unroll
    for (int ct = 0; ct < 4; ++ct) {
#pragma unroll
      for (int kk = 0; kk < 2; ++kk) {
        bf16x16 af = load_frag_a(pbase + kk * 32, 64);
        bf16x16 bf = load_frag_b(&Vt[(ct * 16) * 64] + kk * 32, 64);
        oc[ct] = wmma_bf16(af, bf, oc[ct]);
      }
    }
  }
  __syncthreads();

  // normalize and emit context (bf16) for the out-projection GEMM
  float inv[8];
#pragma unroll
  for (int r = 0; r < 8; ++r) inv[r] = 1.f / l_s[wave * 16 + hf * 8 + r];
#pragma unroll
  for (int ct = 0; ct < 4; ++ct)
#pragma unroll
    for (int r = 0; r < 8; ++r) {
      size_t row = qrow0 + wave * 16 + hf * 8 + r;
      ctx[row * Esz + hc + ct * 16 + nl] = (__bf16)(oc[ct][r] * inv[r]);
    }
}

// ---- residual add + LayerNorm, in place on d_out part 1 -------------------
__global__ __launch_bounds__(256)
void k_add_ln(float* __restrict__ io, const float* __restrict__ res,
              const float* __restrict__ g, const float* __restrict__ bb) {
  __shared__ float r1[256], r2[256];
  const size_t row = blockIdx.x;
  const int tid = threadIdx.x;
  float x[4], s = 0.f, s2 = 0.f;
#pragma unroll
  for (int i = 0; i < 4; ++i) {
    int c = i * 256 + tid;
    x[i] = io[row * Esz + c] + res[row * Esz + c];
    s += x[i]; s2 += x[i] * x[i];
  }
  r1[tid] = s; r2[tid] = s2; __syncthreads();
  for (int off = 128; off; off >>= 1) {
    if (tid < off) { r1[tid] += r1[tid + off]; r2[tid] += r2[tid + off]; }
    __syncthreads();
  }
  float mu = r1[0] * (1.f / Esz);
  float var = r2[0] * (1.f / Esz) - mu * mu;
  float ri = rsqrtf(var + LNEPS);
#pragma unroll
  for (int i = 0; i < 4; ++i) {
    int c = i * 256 + tid;
    io[row * Esz + c] = (x[i] - mu) * ri * g[c] + bb[c];
  }
}

// ---- gate + blend + LayerNorm -> new_mem ----------------------------------
__global__ __launch_bounds__(256)
void k_newmem(const float* __restrict__ gate_mem, const float* __restrict__ gate_upd,
              const float* __restrict__ upd, const float* __restrict__ mem,
              const float* __restrict__ g, const float* __restrict__ bb,
              float* __restrict__ out) {
  __shared__ float r1[256], r2[256];
  const int row = blockIdx.x;                // b*512 + m
  const int b = row >> 9, m = row & 511;
  const int tid = threadIdx.x;
  float x[4], s = 0.f, s2 = 0.f;
#pragma unroll
  for (int i = 0; i < 4; ++i) {
    int c = i * 256 + tid;
    float t  = gate_mem[(size_t)m * Esz + c] + gate_upd[(size_t)b * Esz + c];
    float gt = 1.f / (1.f + __expf(-t));
    float u  = upd[(size_t)b * Esz + c];
    float mv = mem[(size_t)m * Esz + c];
    x[i] = gt * u + (1.f - gt) * mv;
    s += x[i]; s2 += x[i] * x[i];
  }
  r1[tid] = s; r2[tid] = s2; __syncthreads();
  for (int off = 128; off; off >>= 1) {
    if (tid < off) { r1[tid] += r1[tid + off]; r2[tid] += r2[tid + off]; }
    __syncthreads();
  }
  float mu = r1[0] * (1.f / Esz);
  float var = r2[0] * (1.f / Esz) - mu * mu;
  float ri = rsqrtf(var + LNEPS);
#pragma unroll
  for (int i = 0; i < 4; ++i) {
    int c = i * 256 + tid;
    out[(size_t)row * Esz + c] = (x[i] - mu) * ri * g[c] + bb[c];
  }
}

// ---------------------------------------------------------------------------

extern "C" void kernel_launch(void* const* d_in, const int* in_sizes, int n_in,
                              void* d_out, int out_size, void* d_ws, size_t ws_size,
                              hipStream_t stream) {
  const float* query   = (const float*)d_in[0];
  const float* memory  = (const float*)d_in[1];
  const float* r_in_w  = (const float*)d_in[2];
  const float* r_in_b  = (const float*)d_in[3];
  const float* r_out_w = (const float*)d_in[4];
  const float* r_out_b = (const float*)d_in[5];
  const float* w_in_w  = (const float*)d_in[6];
  const float* w_in_b  = (const float*)d_in[7];
  const float* w_out_w = (const float*)d_in[8];
  const float* w_out_b = (const float*)d_in[9];
  const float* rn_g    = (const float*)d_in[10];
  const float* rn_b    = (const float*)d_in[11];
  const float* wn_g    = (const float*)d_in[12];
  const float* wn_b    = (const float*)d_in[13];
  const float* gate_w  = (const float*)d_in[14];
  const float* gate_b  = (const float*)d_in[15];

  // ---- workspace carve-up (all 256B aligned) ----
  char* wsp = (char*)d_ws;
  auto carve = [&](size_t bytes) -> char* {
    char* p = wsp; wsp += (bytes + 255) & ~(size_t)255; return p;
  };
  __bf16* Wq_bf  = (__bf16*)carve((size_t)Esz * Esz * 2);
  __bf16* Wk_bf  = (__bf16*)carve((size_t)Esz * Esz * 2);
  __bf16* Wv_bf  = (__bf16*)carve((size_t)Esz * Esz * 2);
  __bf16* Wo_bf  = (__bf16*)carve((size_t)Esz * Esz * 2);
  __bf16* Wg1_bf = (__bf16*)carve((size_t)Esz * Esz * 2);
  float*  wq     = (float*) carve((size_t)Bsz * Esz * 4);
  float*  vw     = (float*) carve((size_t)Bsz * Esz * 4);
  float*  upd    = (float*) carve((size_t)Bsz * Esz * 4);
  float*  gupd   = (float*) carve((size_t)Bsz * Esz * 4);
  __bf16* Kb     = (__bf16*)carve((size_t)Msz * Esz * 2);
  __bf16* Vb     = (__bf16*)carve((size_t)Msz * Esz * 2);
  float*  gmem   = (float*) carve((size_t)Msz * Esz * 4);
  __bf16* Qp     = (__bf16*)carve((size_t)Bsz * Ssz * Esz * 2);
  __bf16* ctx    = (__bf16*)carve((size_t)Bsz * Ssz * Esz * 2);

  float* out1 = (float*)d_out;                         // (B,S,E) mem_out
  float* out2 = out1 + (size_t)Bsz * Ssz * Esz;        // (B,M,E) new_mem

  const int CT = Esz * Esz;                            // 1M elements

  // 1) weight conversions fp32 -> bf16
  k_convert<<<CT / 256, 256, 0, stream>>>(r_in_w,                Wq_bf,  Esz, Esz,     CT);
  k_convert<<<CT / 256, 256, 0, stream>>>(r_in_w + (size_t)CT,   Wk_bf,  Esz, Esz,     CT);
  k_convert<<<CT / 256, 256, 0, stream>>>(r_in_w + 2*(size_t)CT, Wv_bf,  Esz, Esz,     CT);
  k_convert<<<CT / 256, 256, 0, stream>>>(r_out_w,               Wo_bf,  Esz, Esz,     CT);
  k_convert<<<CT / 256, 256, 0, stream>>>(gate_w,                Wg1_bf, Esz, 2 * Esz, CT);

  // 2) write-MHA collapsed path (softmax over 1 key == identity)
  k_wq<<<(Bsz * Esz) / 256, 256, 0, stream>>>(query, wq);
  k_gemv<<<(Bsz * Esz) / 256, 256, 0, stream>>>(wq,  w_in_w + 2*(size_t)CT, Esz,
                                                w_in_b + 2 * Esz, vw,  Bsz, Esz, Esz);
  k_gemv<<<(Bsz * Esz) / 256, 256, 0, stream>>>(vw,  w_out_w, Esz, w_out_b,
                                                upd, Bsz, Esz, Esz);
  k_gemv<<<(Bsz * Esz) / 256, 256, 0, stream>>>(upd, gate_w + Esz, 2 * Esz, gate_b,
                                                gupd, Bsz, Esz, Esz);

  // 3) projections (WMMA GEMMs, W tiles via TDM)
  k_gemm<float, true><<<dim3(Esz / 64, (Bsz * Ssz) / 128), 256, 0, stream>>>(
      query, Esz, Wq_bf, Esz, r_in_b, Qp, Esz, Esz, 0.125f);
  k_gemm<float, true><<<dim3(Esz / 64, Msz / 128), 256, 0, stream>>>(
      memory, Esz, Wk_bf, Esz, r_in_b + Esz, Kb, Esz, Esz, 1.0f);
  k_gemm<float, true><<<dim3(Esz / 64, Msz / 128), 256, 0, stream>>>(
      memory, Esz, Wv_bf, Esz, r_in_b + 2 * Esz, Vb, Esz, Esz, 1.0f);
  k_gemm<float, false><<<dim3(Esz / 64, Msz / 128), 256, 0, stream>>>(
      memory, Esz, Wg1_bf, Esz, (const float*)nullptr, gmem, Esz, Esz, 1.0f);

  // 4) attention (flash-style, WMMA for QK^T and P@V, TDM-staged Q/K)
  k_attn<<<dim3(Ssz / 64, Hsz, Bsz), 128, 0, stream>>>(Qp, Kb, Vb, ctx);

  // 5) output projection straight into d_out part 1 (fp32)
  k_gemm<__bf16, false><<<dim3(Esz / 64, (Bsz * Ssz) / 128), 256, 0, stream>>>(
      ctx, Esz, Wo_bf, Esz, r_out_b, out1, Esz, Esz, 1.0f);

  // 6) residual + LayerNorm (in place)
  k_add_ln<<<Bsz * Ssz, 256, 0, stream>>>(out1, query, rn_g, rn_b);

  // 7) gate + blend + LayerNorm -> new_mem
  k_newmem<<<Bsz * Msz, 256, 0, stream>>>(gmem, gupd, upd, memory, wn_g, wn_b, out2);
}
